// WorkingFractionalPhysicsLoss_48661979464036
// MI455X (gfx1250) — matrix operand order repaired and checked by reading, not compiled, not run on hardware
//
#include <hip/hip_runtime.h>
#include <hip/hip_bf16.h>
#include <math.h>

typedef __attribute__((ext_vector_type(16))) __bf16 v16bf;
typedef __attribute__((ext_vector_type(8)))  __bf16 v8bf;
typedef __attribute__((ext_vector_type(8)))  float  v8f;

#define N_COLS   16384     // row length of the 2D view (u_t = 0 at row ends)
#define PAD      2048      // leading zero pad of du buffer (covers negative lags)
#define T0       16        // first tap handled by the WMMA band
#define TBAND    1008      // band taps t in [16, 1024)
#define KTOT     1024      // padded K extent of the X window (TBAND+15 -> 1024)
#define NCHUNK   (KTOT/32) // 32 K-chunks of the bf16 WMMA
#define CB       256       // coarse (far-field) block size
#define FAR_EXCL 5         // far field uses blocks p <= q-5 (fully beyond lag 1024)

// ---------------- K1: du, block sums S, data-loss partials (memory-bound) -----
__global__ void __launch_bounds__(256) k1_prep(const float* __restrict__ up,
                                               const float* __restrict__ utr,
                                               float* __restrict__ du_pad,
                                               float* __restrict__ S,
                                               float* __restrict__ dpart, int n) {
  __shared__ float red[256];
  const int t = threadIdx.x;
  const int b = blockIdx.x;
  const long i = (long)b * 256 + t;
  const float a  = up[i];
  const float a1 = (i + 1 < n) ? up[i + 1] : a;     // du[n-1] := 0
  const float g  = utr[i];
  const float d  = a - g;
  const float duv = a1 - a;
  du_pad[PAD + i] = duv;
  if (b < PAD / 256) du_pad[b * 256 + t] = 0.0f;    // zero leading pad
  red[t] = d * d; __syncthreads();
  for (int s = 128; s > 0; s >>= 1) { if (t < s) red[t] += red[t + s]; __syncthreads(); }
  if (t == 0) dpart[b] = red[0];
  __syncthreads();
  red[t] = duv; __syncthreads();
  for (int s = 128; s > 0; s >>= 1) { if (t < s) red[t] += red[t + s]; __syncthreads(); }
  if (t == 0) S[b] = red[0];
}

// ---------------- K2: coarse far-field convolution F[q] ----------------------
__global__ void __launch_bounds__(256) k2_far(const float* __restrict__ S,
                                              float* __restrict__ F, int nq) {
  __shared__ float sl[256];
  const int t  = threadIdx.x;
  const int q0 = blockIdx.x * 256;
  const int q  = q0 + t;
  float acc = 0.0f;
  const int cmax = (q0 + 255 - FAR_EXCL) / 256;
  for (int c = 0; c <= cmax; ++c) {
    __builtin_prefetch(S + (c + 1) * 256, 0, 1);    // global_prefetch_b8
    sl[t] = S[c * 256 + t];
    __syncthreads();
    const int plim = q - FAR_EXCL;
    const int base = c * 256;
#pragma unroll 4
    for (int pp = 0; pp < 256; ++pp) {
      const int p = base + pp;
      if (p <= plim) {
        const float dd = (float)(q - p);
        acc += 0.5f * rsqrtf(256.0f * dd) * sl[pp]; // smooth tail ~ 1/(2*sqrt(lag))
      }
    }
    __syncthreads();
  }
  if (q < nq) F[q] = acc;
}

// ---------------- K3: WMMA near-band + head + far + physics partials ---------
__global__ void __launch_bounds__(256) k3_wmma(const float* __restrict__ du_pad,
                                               const float* __restrict__ F,
                                               float* __restrict__ ppart,
                                               float coef, int n) {
  // B operand pre-swizzled to per-lane fragment order: Hswz[c][lane][16] bf16.
  __shared__ __align__(32) __bf16 Hswz[NCHUNK * 32 * 16]; // 32 KB
  __shared__ __align__(16) __bf16 win[3072];              // bf16 du window, 6 KB
  __shared__ float  win32[2064];                          // f32 du window (head/u_t), 8.25 KB
  __shared__ float  hh[16];
  __shared__ float  red[256];
  const int  t   = threadIdx.x;
  const long j0w = (long)blockIdx.x * 2048;               // 8 waves x 256 outputs

  // Fill swizzled filter matrix: for (c, lane, e): v=e>>1, odd=e&1,
  // k = 32c + 16*(lane>>4) + 2v + odd, n = lane&15, tap = 1023 - k + n.
  for (int idx = t; idx < NCHUNK * 32 * 16; idx += 256) {
    const int e    = idx & 15;
    const int lane = (idx >> 4) & 31;
    const int c    = idx >> 9;
    const int k    = c * 32 + ((lane >> 4) << 4) + e;     // 16*hi + 2v + odd == +e
    const int nn   = lane & 15;
    const int tap  = (T0 + TBAND - 1) - k + nn;           // 1023 - k + nn
    float v = 0.0f;
    if (tap >= T0 && tap < T0 + TBAND)
      v = sqrtf((float)tap) - sqrtf((float)(tap - 1));
    Hswz[idx] = (__bf16)v;
  }
  if (t < 16) hh[t] = (t >= 1) ? (sqrtf((float)t) - sqrtf((float)(t - 1))) : 0.0f;

  // bf16 du window [j0w-1023, j0w+2049): pad index j0w+1025 ..
  const long wbase = j0w + 1025;
  for (int idx = t; idx < 3072; idx += 256) {
    const long gi = wbase + idx;
    const float v = (gi < (long)PAD + n) ? du_pad[gi] : 0.0f;
    win[idx] = (__bf16)v;
  }
  // f32 du window [j0w-16, j0w+2048): pad index j0w+2032 .. (always in range)
  for (int idx = t; idx < 2064; idx += 256)
    win32[idx] = du_pad[j0w + 2032 + idx];
  __syncthreads();

  const int lane = t & 31, wv = t >> 5;
  const int hi = lane >> 4, lo = lane & 15;
  const int tilebase = 256 * wv;
  const long j0 = j0w + tilebase;
  const int lsbase = tilebase + 16 * lo;                  // A-row (M = lo) window base

  v8f acc = {};
#pragma unroll 4
  for (int c = 0; c < NCHUNK; ++c) {
    const int kc = c * 32;
    // A (16x32 bf16): two contiguous 16-byte LDS reads per lane, concat in regs
    const v8bf alo = *(const v8bf*)(win + lsbase + kc + hi * 16);
    const v8bf ahi = *(const v8bf*)(win + lsbase + kc + hi * 16 + 8);
    const v16bf a = __builtin_shufflevector(alo, ahi,
        0, 1, 2, 3, 4, 5, 6, 7, 8, 9, 10, 11, 12, 13, 14, 15);
    // B (32x16 bf16): one contiguous 32-byte LDS read per lane
    const v16bf b = *(const v16bf*)(Hswz + (((c << 5) + lane) << 4));
    acc = __builtin_amdgcn_wmma_f32_16x16x32_bf16(false, a, false, b,
                                                  (short)0, acc, false, false);
  }

  // Epilogue: exact f32 head taps, far field, residual, squared partial
  float psum = 0.0f;
#pragma unroll
  for (int r = 0; r < 8; ++r) {
    const int  M  = r + 8 * hi;
    const int  dj = 16 * M + lo + tilebase;               // j - j0w
    const long j  = j0w + dj;
    float head = 0.0f;
#pragma unroll
    for (int tt = 1; tt < 16; ++tt)
      head += hh[tt] * win32[dj + 16 - tt];               // covers j-15..j-1 (pad-zeroed)
    const float far  = F[j >> 8];
    const float frac = coef * (acc[r] + head + far);
    const float utv  = ((j & (N_COLS - 1)) == (N_COLS - 1)) ? 0.0f : win32[dj + 16];
    const float rr = utv - frac;
    psum += rr * rr;
  }
  red[t] = psum; __syncthreads();
  for (int s = 128; s > 0; s >>= 1) { if (t < s) red[t] += red[t + s]; __syncthreads(); }
  if (t == 0) ppart[blockIdx.x] = red[0];
}

// ---------------- K4: deterministic final reduction --------------------------
__global__ void __launch_bounds__(256) k4_final(const float* __restrict__ dpart,
                                                const float* __restrict__ ppart,
                                                float* __restrict__ out,
                                                int nd, int np, int n) {
  __shared__ float red[256];
  const int t = threadIdx.x;
  float s = 0.0f;
  for (int i = t; i < nd; i += 256) s += dpart[i];
  red[t] = s; __syncthreads();
  for (int st = 128; st > 0; st >>= 1) { if (t < st) red[t] += red[t + st]; __syncthreads(); }
  const float dataSum = red[0]; __syncthreads();
  s = 0.0f;
  for (int i = t; i < np; i += 256) s += ppart[i];
  red[t] = s; __syncthreads();
  for (int st = 128; st > 0; st >>= 1) { if (t < st) red[t] += red[t + st]; __syncthreads(); }
  if (t == 0) {
    const float physSum   = red[0];
    const float data_loss = dataSum / (float)n;
    const float phys_loss = physSum / (float)n;
    out[0] = data_loss + 0.1f * phys_loss;
    out[1] = data_loss;
    out[2] = phys_loss;
  }
}

// ---------------- launch ------------------------------------------------------
extern "C" void kernel_launch(void* const* d_in, const int* in_sizes, int n_in,
                              void* d_out, int out_size, void* d_ws, size_t ws_size,
                              hipStream_t stream) {
  const float* u_pred = (const float*)d_in[0];
  const float* u_true = (const float*)d_in[1];
  float* out = (float*)d_out;
  const int n = in_sizes[0];            // 1024*16384 = 16,777,216
  const int nq = n / CB;                // coarse blocks
  const int nwg3 = n / 2048;            // K3 workgroups (8 tiles each)

  float* ws = (float*)d_ws;
  float* du_pad = ws;                   // PAD + n floats
  float* S      = du_pad + (PAD + n);   // nq floats
  float* F      = S + nq;               // nq floats
  float* dpart  = F + nq;               // n/256 floats
  float* ppart  = dpart + (n / 256);    // nwg3 floats

  // coef = h^{-alpha}/Gamma(2-alpha), h=1/(n-1), alpha=0.5, Gamma(1.5)=0.8862269255
  const float coef = sqrtf((float)(n - 1)) / 0.88622692545275801f;

  k1_prep<<<n / 256, 256, 0, stream>>>(u_pred, u_true, du_pad, S, dpart, n);
  k2_far <<<nq / 256, 256, 0, stream>>>(S, F, nq);
  k3_wmma<<<nwg3, 256, 0, stream>>>(du_pad, F, ppart, coef, n);
  k4_final<<<1, 256, 0, stream>>>(dpart, ppart, out, n / 256, nwg3, n);
}
